// GNNModel_3143916060989
// MI455X (gfx1250) — matrix operand order repaired and checked by reading, not compiled, not run on hardware
//
#include <hip/hip_runtime.h>
#include <hip/hip_bf16.h>
#include <math.h>

#define H         128
#define LAYERS    3
#define GD        32
#define NEG_SLOPE 0.2f
#define GEPS      1e-5f

typedef __attribute__((ext_vector_type(16))) __bf16 v16bf;
typedef __attribute__((ext_vector_type(8)))  float  v8f;

__device__ __forceinline__ unsigned short f2bf(float f) {
    unsigned u = __float_as_uint(f);
    unsigned r = ((u >> 16) & 1u) + 0x7FFFu;      // round-to-nearest-even
    return (unsigned short)((u + r) >> 16);
}

// ---------------------------------------------------------------------------
// Dual GEMM: XL = X @ Wsrc, XR = X @ Wdst  (bf16 WMMA, fp32 accumulate)
// Block: 256 threads (8 waves), 128 rows x 128 cols.
//  - X tile is brought into LDS with GLOBAL_LOAD_ASYNC_TO_LDS_B128 (ASYNCcnt),
//    overlapped with the weight staging, then converted to bf16 fragments.
//  - LDS fragment buffers use the exact ISA WMMA layouts so each fragment
//    load is two contiguous ds_load_b128 per lane.
// ---------------------------------------------------------------------------
__launch_bounds__(256)
__global__ void gemm_dual_kernel(const float* Xin,
                                 const float* Wsrc,    // [H][H] this layer
                                 const float* Wdst,
                                 float* XL, float* XR, int N) {
    __shared__ __align__(16) float          lX [128 * H];   // raw X tile (64KB)
    __shared__ __align__(32) unsigned short lA [32 * 512];  // 8 mt * 4 ks frags
    __shared__ __align__(32) unsigned short lBs[32 * 512];  // 4 ks * 8 nt frags
    __shared__ __align__(32) unsigned short lBd[32 * 512];

    const int tid     = threadIdx.x;
    const int rowBase = blockIdx.x * 128;

    // ---- 1) fire async copy of the raw X tile into LDS (16B per lane/round)
    {
        unsigned ldsBase = (unsigned)(size_t)&lX[0];       // low 32b = LDS offset
        const float* gbase = Xin + (size_t)rowBase * H;
#pragma unroll
        for (int i = 0; i < 16; ++i) {
            int elt = (i * 256 + tid) * 4;                 // float index (x4/lane)
            int row = elt >> 7;
            unsigned ldsOff = ldsBase + (unsigned)(elt * 4);
            unsigned gOff   = (unsigned)(elt * 4);
            if (rowBase + row < N) {
                asm volatile(
                    "global_load_async_to_lds_b128 %0, %1, %2 offset:0"
                    :: "v"(ldsOff), "v"(gOff), "s"(gbase)
                    : "memory");
            }
        }
    }

    // ---- 2) stage weights in B-fragment layout (overlaps the async copy):
    // B 32x16 16-bit tile: lane = n + ((k>>3)&1)*16, element = (k&7)+(k>=16?8:0).
    for (int t = tid; t < H * H; t += 256) {
        int k = t >> 7, n = t & 127;                   // W[k][n]
        int frag = ((k >> 5) << 3) + (n >> 4);         // (ks, nt)
        int kk = k & 31, nn = n & 15;
        int lane = nn + (((kk >> 3) & 1) << 4);
        int e    = (kk & 7) + ((kk >= 16) ? 8 : 0);
        int o = frag * 512 + lane * 16 + e;
        lBs[o] = f2bf(Wsrc[t]);
        lBd[o] = f2bf(Wdst[t]);
    }

    // ---- 3) wait for the tensor-style async copy, make it block-visible
    asm volatile("s_wait_asynccnt 0" ::: "memory");
    __syncthreads();

    // ---- 4) convert raw X tile -> bf16 A-fragment layout:
    // A 16x32 16-bit tile: lane = m + ((k>>3)&1)*16, elem = (k&7)+(k>=16?8:0).
    for (int t = tid; t < 128 * H; t += 256) {
        int r = t >> 7, c = t & 127;
        float v = (rowBase + r < N) ? lX[t] : 0.0f;
        int mt = r >> 4, m = r & 15;
        int ks = c >> 5, k = c & 31;
        int frag = mt * 4 + ks;
        int lane = m + (((k >> 3) & 1) << 4);
        int e    = (k & 7) + ((k >= 16) ? 8 : 0);
        lA[frag * 512 + lane * 16 + e] = f2bf(v);
    }
    __syncthreads();

    // ---- 5) WMMA: each wave owns one 16-row strip, both output matrices
    const int w    = tid >> 5;       // wave id == row-tile mt (0..7)
    const int lane = tid & 31;

    v8f zero = {0.f, 0.f, 0.f, 0.f, 0.f, 0.f, 0.f, 0.f};
    v8f accL[8], accR[8];
#pragma unroll
    for (int nt = 0; nt < 8; ++nt) { accL[nt] = zero; accR[nt] = zero; }

#pragma unroll
    for (int ks = 0; ks < 4; ++ks) {
        const v16bf a = *(const v16bf*)&lA[(w * 4 + ks) * 512 + lane * 16];
#pragma unroll
        for (int nt = 0; nt < 8; ++nt) {
            const v16bf bs = *(const v16bf*)&lBs[(ks * 8 + nt) * 512 + lane * 16];
            accL[nt] = __builtin_amdgcn_wmma_f32_16x16x32_bf16(
                false, a, false, bs, (short)0, accL[nt], false, false);
            const v16bf bd = *(const v16bf*)&lBd[(ks * 8 + nt) * 512 + lane * 16];
            accR[nt] = __builtin_amdgcn_wmma_f32_16x16x32_bf16(
                false, a, false, bd, (short)0, accR[nt], false, false);
        }
    }

    // C/D layout: vgpr r -> row M = r + (lane>=16 ? 8 : 0), col N = lane&15.
    const int nc   = lane & 15;
    const int mOff = (lane >= 16) ? 8 : 0;
#pragma unroll
    for (int nt = 0; nt < 8; ++nt) {
#pragma unroll
        for (int r = 0; r < 8; ++r) {
            int row = rowBase + w * 16 + r + mOff;
            if (row < N) {
                int col = nt * 16 + nc;
                XL[row * H + col] = accL[nt][r];
                XR[row * H + col] = accR[nt][r];
            }
        }
    }
}

// ---------------------------------------------------------------------------
// Edge attention logits + segment max (order-preserving uint encoding).
// One wave32 per edge; each lane handles a float4 (coalesced 512B row reads).
// ---------------------------------------------------------------------------
__launch_bounds__(256)
__global__ void edge_logit_kernel(const float* XL, const float* XR,
                                  const float* att, const int* ei,
                                  float* ebuf, unsigned int* emaxU,
                                  int E, int ET) {
    int wid = blockIdx.x * (blockDim.x >> 5) + (threadIdx.x >> 5);
    if (wid >= ET) return;
    int lane = threadIdx.x & 31;
    int s, d;
    if (wid < E) { s = ei[wid]; d = ei[E + wid]; }
    else         { s = wid - E; d = s; }               // self loop

    const float4 a = *(const float4*)&XL[s * H + lane * 4];
    const float4 b = *(const float4*)&XR[d * H + lane * 4];
    const float4 w = *(const float4*)&att[lane * 4];

    float sum = 0.f, v;
    v = a.x + b.x; v = v > 0.f ? v : NEG_SLOPE * v; sum += v * w.x;
    v = a.y + b.y; v = v > 0.f ? v : NEG_SLOPE * v; sum += v * w.y;
    v = a.z + b.z; v = v > 0.f ? v : NEG_SLOPE * v; sum += v * w.z;
    v = a.w + b.w; v = v > 0.f ? v : NEG_SLOPE * v; sum += v * w.w;
#pragma unroll
    for (int off = 16; off; off >>= 1) sum += __shfl_xor(sum, off, 32);

    if (lane == 0) {
        ebuf[wid] = sum;
        unsigned bits = __float_as_uint(sum);
        unsigned enc  = (bits & 0x80000000u) ? ~bits : (bits | 0x80000000u);
        atomicMax(&emaxU[d], enc);
    }
}

// exp(e - max) and per-destination denominator
__global__ void edge_exp_kernel(const int* ei, float* ebuf,
                                const unsigned int* emaxU, float* denom,
                                int E, int ET) {
    int i = blockIdx.x * blockDim.x + threadIdx.x;
    if (i >= ET) return;
    int d = (i < E) ? ei[E + i] : (i - E);
    unsigned u    = emaxU[d];
    unsigned bits = (u & 0x80000000u) ? (u ^ 0x80000000u) : ~u;
    float m  = __uint_as_float(bits);
    float ex = __expf(ebuf[i] - m);
    ebuf[i]  = ex;
    atomicAdd(&denom[d], ex);
}

// h = b_conv broadcast (init before scatter-add)
__global__ void h_init_kernel(float* h, const float* bconv, int total) {
    int i = blockIdx.x * blockDim.x + threadIdx.x;
    if (i < total) h[i] = bconv[i & (H - 1)];
}

// h[dst] += alpha * XL[src]; wave per edge, lane per float4, f32 atomics (L2)
__launch_bounds__(256)
__global__ void edge_aggr_kernel(const float* XL, const int* ei,
                                 const float* ebuf, const float* denom,
                                 float* h, int E, int ET) {
    int wid = blockIdx.x * (blockDim.x >> 5) + (threadIdx.x >> 5);
    if (wid >= ET) return;
    int lane = threadIdx.x & 31;
    int s, d;
    if (wid < E) { s = ei[wid]; d = ei[E + wid]; }
    else         { s = wid - E; d = s; }
    float alpha = ebuf[wid] / denom[d];
    const float4 a = *(const float4*)&XL[s * H + lane * 4];
    float* hp = &h[d * H + lane * 4];
    atomicAdd(hp + 0, alpha * a.x);
    atomicAdd(hp + 1, alpha * a.y);
    atomicAdd(hp + 2, alpha * a.z);
    atomicAdd(hp + 3, alpha * a.w);
}

__global__ void counts_kernel(const int* batch, float* counts, int N) {
    int i = blockIdx.x * blockDim.x + threadIdx.x;
    if (i < N) atomicAdd(&counts[batch[i]], 1.0f);
}

// Segment sum over sorted `batch`: block covers 64 nodes; 2 threads/column,
// run-length accumulate locally, flush one atomic per graph run.
__launch_bounds__(256)
__global__ void seg_sum_kernel(const float* in, const int* batch,
                               float* out, int N) {
    int col  = threadIdx.x & (H - 1);
    int half = threadIdx.x >> 7;
    int base = blockIdx.x * 64 + half * 32;
    float acc = 0.f; int curg = -1;
    for (int r = 0; r < 32; ++r) {
        int row = base + r;
        if (row >= N) break;
        int g = batch[row];
        if (g != curg) {
            if (curg >= 0) atomicAdd(&out[curg * H + col], acc);
            curg = g; acc = 0.f;
        }
        acc += in[row * H + col];
    }
    if (curg >= 0) atomicAdd(&out[curg * H + col], acc);
}

// Segment sum of (h - ms*mean)^2
__launch_bounds__(256)
__global__ void seg_var_kernel(const float* h, const int* batch,
                               const float* meanBuf, const float* counts,
                               const float* gnms, float* varBuf, int N) {
    int col  = threadIdx.x & (H - 1);
    int half = threadIdx.x >> 7;
    int base = blockIdx.x * 64 + half * 32;
    float ms  = gnms[col];
    float acc = 0.f, mcur = 0.f; int curg = -1;
    for (int r = 0; r < 32; ++r) {
        int row = base + r;
        if (row >= N) break;
        int g = batch[row];
        if (g != curg) {
            if (curg >= 0) atomicAdd(&varBuf[curg * H + col], acc);
            curg = g; acc = 0.f;
            mcur = ms * meanBuf[g * H + col] / fmaxf(counts[g], 1.f);
        }
        float sub = h[row * H + col] - mcur;
        acc += sub * sub;
    }
    if (curg >= 0) atomicAdd(&varBuf[curg * H + col], acc);
}

// GraphNorm affine + ReLU -> next layer input
__global__ void gnorm_relu_kernel(const float* h, const int* batch,
                                  const float* meanBuf, const float* varBuf,
                                  const float* counts,
                                  const float* gnw, const float* gnb,
                                  const float* gnms,
                                  float* xout, int total) {
    int i = blockIdx.x * blockDim.x + threadIdx.x;
    if (i >= total) return;
    int row = i >> 7, col = i & (H - 1);
    int g = batch[row];
    float cnt  = fmaxf(counts[g], 1.f);
    float mean = meanBuf[g * H + col] / cnt;
    float var  = varBuf[g * H + col] / cnt;
    float sub  = h[i] - gnms[col] * mean;
    float y    = gnw[col] * sub * rsqrtf(var + GEPS) + gnb[col];
    xout[i]    = fmaxf(y, 0.f);
}

// Final MLP head on [G, 3H+GD]: z = relu(hcat@W1+b1); out = z@W2+b2
__launch_bounds__(128)
__global__ void mlp_kernel(const float* poolBuf, const float* counts,
                           const float* gf, const float* W1, const float* b1,
                           const float* W2, const float* b2,
                           float* out, int G) {
    __shared__ float z[H];
    int g = blockIdx.x;
    int j = threadIdx.x;
    float inv = 1.f / fmaxf(counts[g], 1.f);
    float acc = b1[j];
    for (int l = 0; l < LAYERS; ++l) {
        const float* p = &poolBuf[(l * G + g) * H];
        for (int i = 0; i < H; ++i)
            acc += (p[i] * inv) * W1[(l * H + i) * H + j];
    }
    const float* gfg = &gf[g * GD];
    for (int i = 0; i < GD; ++i)
        acc += gfg[i] * W1[(LAYERS * H + i) * H + j];
    z[j] = fmaxf(acc, 0.f) * W2[j];
    __syncthreads();
    for (int s = 64; s > 0; s >>= 1) {
        if (j < s) z[j] += z[j + s];
        __syncthreads();
    }
    if (j == 0) out[g] = z[0] + b2[0];
}

// ---------------------------------------------------------------------------
extern "C" void kernel_launch(void* const* d_in, const int* in_sizes, int n_in,
                              void* d_out, int out_size, void* d_ws, size_t ws_size,
                              hipStream_t stream) {
    (void)n_in; (void)out_size; (void)ws_size;
    const float* x     = (const float*)d_in[0];
    const float* gf    = (const float*)d_in[1];
    const float* Wsrc  = (const float*)d_in[2];
    const float* Wdst  = (const float*)d_in[3];
    const float* att   = (const float*)d_in[4];
    const float* bconv = (const float*)d_in[5];
    const float* gnw   = (const float*)d_in[6];
    const float* gnb   = (const float*)d_in[7];
    const float* gnms  = (const float*)d_in[8];
    const float* W1    = (const float*)d_in[9];
    const float* b1    = (const float*)d_in[10];
    const float* W2    = (const float*)d_in[11];
    const float* b2    = (const float*)d_in[12];
    const int*   ei    = (const int*)d_in[13];
    const int*   batch = (const int*)d_in[14];

    const int N  = in_sizes[0] / H;
    const int G  = in_sizes[1] / GD;
    const int E  = in_sizes[13] / 2;
    const int ET = E + N;

    char* ws = (char*)d_ws;
    size_t off = 0;
    auto alloc = [&](size_t bytes) -> void* {
        void* p = ws + off;
        off += (bytes + 255) & ~(size_t)255;
        return p;
    };
    float*        XL      = (float*)alloc((size_t)N * H * 4);
    float*        XR      = (float*)alloc((size_t)N * H * 4);   // also x_next
    float*        hbuf    = (float*)alloc((size_t)N * H * 4);
    float*        ebuf    = (float*)alloc((size_t)ET * 4);
    unsigned int* emaxU   = (unsigned int*)alloc((size_t)N * 4);
    float*        denom   = (float*)alloc((size_t)N * 4);
    float*        counts  = (float*)alloc((size_t)G * 4);
    float*        meanBuf = (float*)alloc((size_t)G * H * 4);
    float*        varBuf  = (float*)alloc((size_t)G * H * 4);
    float*        poolBuf = (float*)alloc((size_t)LAYERS * G * H * 4);
    float*        xnext   = XR;  // XR dead after logit pass; safe alias

    hipMemsetAsync(counts, 0, (size_t)G * 4, stream);
    counts_kernel<<<(N + 255) / 256, 256, 0, stream>>>(batch, counts, N);

    const float* xin = x;
    const int rowBlocks  = (N + 127) / 128;
    const int waveBlocks = (ET + 7) / 8;          // 8 waves / 256-thread block
    const int elemBlocks = (N * H + 255) / 256;
    for (int l = 0; l < LAYERS; ++l) {
        gemm_dual_kernel<<<rowBlocks, 256, 0, stream>>>(
            xin, Wsrc + (size_t)l * H * H, Wdst + (size_t)l * H * H, XL, XR, N);

        hipMemsetAsync(emaxU, 0, (size_t)N * 4, stream);
        edge_logit_kernel<<<waveBlocks, 256, 0, stream>>>(
            XL, XR, att + l * H, ei, ebuf, emaxU, E, ET);

        hipMemsetAsync(denom, 0, (size_t)N * 4, stream);
        edge_exp_kernel<<<(ET + 255) / 256, 256, 0, stream>>>(
            ei, ebuf, emaxU, denom, E, ET);

        h_init_kernel<<<elemBlocks, 256, 0, stream>>>(hbuf, bconv + l * H, N * H);
        edge_aggr_kernel<<<waveBlocks, 256, 0, stream>>>(
            XL, ei, ebuf, denom, hbuf, E, ET);

        hipMemsetAsync(meanBuf, 0, (size_t)G * H * 4, stream);
        seg_sum_kernel<<<(N + 63) / 64, 256, 0, stream>>>(hbuf, batch, meanBuf, N);

        hipMemsetAsync(varBuf, 0, (size_t)G * H * 4, stream);
        seg_var_kernel<<<(N + 63) / 64, 256, 0, stream>>>(
            hbuf, batch, meanBuf, counts, gnms + l * H, varBuf, N);

        gnorm_relu_kernel<<<elemBlocks, 256, 0, stream>>>(
            hbuf, batch, meanBuf, varBuf, counts,
            gnw + l * H, gnb + l * H, gnms + l * H, xnext, N * H);

        hipMemsetAsync(poolBuf + (size_t)l * G * H, 0, (size_t)G * H * 4, stream);
        seg_sum_kernel<<<(N + 63) / 64, 256, 0, stream>>>(
            xnext, batch, poolBuf + (size_t)l * G * H, N);

        xin = xnext;
    }
    mlp_kernel<<<G, H, 0, stream>>>(poolBuf, counts, gf, W1, b1, W2, b2,
                                    (float*)d_out, G);
}
